// PolyHashV8H_42606075576734
// MI455X (gfx1250) — matrix operand order repaired and checked by reading, not compiled, not run on hardware
//
#include <hip/hip_runtime.h>
#include <hip/hip_bf16.h>

typedef __attribute__((ext_vector_type(16))) _Float16 v16h;
typedef __attribute__((ext_vector_type(8)))  float    v8f;

#define BM 128
#define BN 64
#define BK 32
#define LDA_PAD 40   // halves per LDS row (80B: 16B-aligned, bank-spread)

#if __has_builtin(__builtin_amdgcn_global_load_async_to_lds_b128) && \
    __has_builtin(__builtin_amdgcn_s_wait_asynccnt)
#define USE_ASYNC_LDS 1
typedef int v4i_b128 __attribute__((vector_size(16)));
typedef __attribute__((address_space(1))) v4i_b128 as1_v4i;
typedef __attribute__((address_space(3))) v4i_b128 as3_v4i;
#else
#define USE_ASYNC_LDS 0
#endif

// ---------------------------------------------------------------------------
// Generic f16 WMMA GEMM:  out = epilogue(A[M,K] @ W[K,N])
//   A row-major f16 (leading dim lda), W row-major f16 (leading dim N)
//   epilogue: +bias, optional SiLU, optional residual add, f32/f16 stores
// Block tile 128x64, 8 waves (wave32), wave tile 16x64 (4 x 16x16 accum).
// A tile staged via async global->LDS copy (ASYNCcnt) when available.
// B tile staged transposed so fragment reads are contiguous b128 loads.
// ---------------------------------------------------------------------------
__global__ __launch_bounds__(256) void gemm_f16_wmma(
    const _Float16* __restrict__ A, int lda,
    const _Float16* __restrict__ W,
    const float* __restrict__ bias,
    const float* __restrict__ res,
    float* __restrict__ outF,
    _Float16* __restrict__ outH,
    int N, int K, int act)
{
    __shared__ _Float16 lsA[BM][LDA_PAD];   // rows of A, K-contiguous
    __shared__ _Float16 lsBT[BN][LDA_PAD];  // B transposed: [n][k]

    const int tid  = threadIdx.x;
    const int wave = tid >> 5;
    const int lane = tid & 31;
    const int m0   = blockIdx.x * BM;
    const int n0   = blockIdx.y * BN;

    union AF { v16h v; _Float16 e[16]; uint4 q[2]; };
    union CF { v8f  v; float    e[8]; };
    union H8 { uint4 q; _Float16 e[8]; };

    CF acc[4];
#pragma unroll
    for (int i = 0; i < 4; ++i)
#pragma unroll
        for (int r = 0; r < 8; ++r) acc[i].e[r] = 0.f;

    const int nKB = K / BK;
    for (int kb = 0; kb < nKB; ++kb) {
        // ---- stage A tile: 128x32 halves, two 16B segments per thread ----
#pragma unroll
        for (int it = 0; it < 2; ++it) {
            int idx = (tid + it * 256) * 8;      // half index 0..4088
            int r = idx >> 5, c = idx & 31;
            const _Float16* src = A + (size_t)(m0 + r) * lda + kb * BK + c;
#if USE_ASYNC_LDS
            __builtin_amdgcn_global_load_async_to_lds_b128(
                (as1_v4i*)src, (as3_v4i*)&lsA[r][c], 0, 0);
#else
            *(uint4*)&lsA[r][c] = *(const uint4*)src;
#endif
        }
        // ---- stage B tile transposed: read 8 halves of a W row, scatter ----
        {
            int idx = tid * 8;                   // 0..2040
            int k = idx >> 6, cb = idx & 63;
            H8 h8;
            h8.q = *(const uint4*)(W + (size_t)(kb * BK + k) * N + n0 + cb);
#pragma unroll
            for (int j = 0; j < 8; ++j) lsBT[cb + j][k] = h8.e[j];
        }
        if (kb + 1 < nKB) {
            __builtin_prefetch(A + (size_t)(m0 + (tid >> 1)) * lda + (kb + 1) * BK, 0, 1);
        }
#if USE_ASYNC_LDS
        __builtin_amdgcn_s_wait_asynccnt(0);
#endif
        __syncthreads();

        // A fragment (ISA 7.12.2): lanes 0-15 rows M, K {0..7,16..23};
        // lanes 16-31 K {8..15,24..31}.  -> two contiguous b128 LDS reads.
        AF fa;
        {
            int row = wave * 16 + (lane & 15);
            int ko  = (lane >= 16) ? 8 : 0;
            fa.q[0] = *(uint4*)&lsA[row][ko];
            fa.q[1] = *(uint4*)&lsA[row][ko + 16];
        }
#pragma unroll
        for (int nt = 0; nt < 4; ++nt) {
            // B fragment: lanes 0-15 K {0..7,16..23}, lanes 16-31 K {8..15,24..31} of column N.
            AF fb;
            int col = nt * 16 + (lane & 15);
            int ko  = (lane >= 16) ? 8 : 0;
            fb.q[0] = *(uint4*)&lsBT[col][ko];
            fb.q[1] = *(uint4*)&lsBT[col][ko + 16];
            acc[nt].v = __builtin_amdgcn_wmma_f32_16x16x32_f16(
                false, fa.v, false, fb.v, (short)0, acc[nt].v, false, false);
        }
        __syncthreads();
    }

    // Epilogue. C/D layout: VGPR r -> row r (+8 for lanes 16-31), col = lane&15.
    const int rbase = m0 + wave * 16 + ((lane >= 16) ? 8 : 0);
#pragma unroll
    for (int nt = 0; nt < 4; ++nt) {
        int col = n0 + nt * 16 + (lane & 15);
        float bv = bias ? bias[col] : 0.f;
#pragma unroll
        for (int r = 0; r < 8; ++r) {
            int row = rbase + r;
            float v = acc[nt].e[r] + bv;
            if (act == 1) v = v / (1.f + __expf(-v));   // SiLU
            if (res)  v += res[(size_t)row * N + col];
            if (outF) outF[(size_t)row * N + col] = v;
            if (outH) outH[(size_t)row * N + col] = (_Float16)v;
        }
    }
}

// ---------------------------------------------------------------------------
// Stage 1: rolling hashes + scale attention -> pre-conv hash feature (B,T,64)
// 257^j mod 4096 == (256*j+1) mod 4096, so hash = sum c[t-j]*(256j+1) mod 4096.
// ---------------------------------------------------------------------------
__global__ __launch_bounds__(256) void featurize_kernel(
    const int* __restrict__ chars, const float* __restrict__ emb_byte,
    const float* __restrict__ hash_tables, const float* __restrict__ Wq,
    const float* __restrict__ bq, float* __restrict__ pre, int T)
{
    int n = blockIdx.x * 256 + threadIdx.x;
    int t = n % T;
    int c[16];
#pragma unroll
    for (int j = 0; j < 16; ++j) c[j] = (t >= j) ? chars[n - j] : 0;

    int acc = 0, h0 = 0, h1 = 0, h2 = 0, h3 = 0;
#pragma unroll
    for (int j = 0; j < 16; ++j) {
        acc += c[j] * ((256 * j + 1) & 4095);
        if (j == 1)  h0 = acc & 4095;
        if (j == 3)  h1 = acc & 4095;
        if (j == 7)  h2 = acc & 4095;
        if (j == 15) h3 = acc & 4095;
    }
    const float* eb = emb_byte + (size_t)c[0] * 64;
    const float* s0 = hash_tables + ((size_t)0 * 4096 + h0) * 64;
    const float* s1 = hash_tables + ((size_t)1 * 4096 + h1) * 64;
    const float* s2 = hash_tables + ((size_t)2 * 4096 + h2) * 64;
    const float* s3 = hash_tables + ((size_t)3 * 4096 + h3) * 64;

    float sc0 = 0.f, sc1 = 0.f, sc2 = 0.f, sc3 = 0.f;
#pragma unroll 1
    for (int e = 0; e < 64; ++e) {
        float q = bq[e];
#pragma unroll 1
        for (int i = 0; i < 64; ++i) q += eb[i] * Wq[i * 64 + e];
        sc0 += q * s0[e]; sc1 += q * s1[e]; sc2 += q * s2[e]; sc3 += q * s3[e];
    }
    sc0 *= 0.125f; sc1 *= 0.125f; sc2 *= 0.125f; sc3 *= 0.125f;  // 1/sqrt(64)
    float mx = fmaxf(fmaxf(sc0, sc1), fmaxf(sc2, sc3));
    float w0 = __expf(sc0 - mx), w1 = __expf(sc1 - mx);
    float w2 = __expf(sc2 - mx), w3 = __expf(sc3 - mx);
    float inv = 1.f / (w0 + w1 + w2 + w3);
    w0 *= inv; w1 *= inv; w2 *= inv; w3 *= inv;

    float* o = pre + (size_t)n * 64;
#pragma unroll 1
    for (int e = 0; e < 64; ++e)
        o[e] = w0 * s0[e] + w1 * s1[e] + w2 * s2[e] + w3 * s3[e];
}

// ---------------------------------------------------------------------------
// Stage 2: causal depthwise conv(K=4)+SiLU, concat features, zero-pad to 160.
// feat row = [byte_emb(64) | conv_silu(64) | match(4) | pad(28)] in f16.
// ---------------------------------------------------------------------------
__global__ __launch_bounds__(256) void build_feat_kernel(
    const int* __restrict__ chars, const float* __restrict__ emb_byte,
    const float* __restrict__ pre, const float* __restrict__ conv_w,
    const float* __restrict__ conv_b, _Float16* __restrict__ feat, int T)
{
    int n = blockIdx.x * 256 + threadIdx.x;
    int t = n % T;
    int ch = chars[n];
    _Float16* o = feat + (size_t)n * 160;
    const float* eb = emb_byte + (size_t)ch * 64;
#pragma unroll 1
    for (int e = 0; e < 64; ++e) o[e] = (_Float16)eb[e];
#pragma unroll 1
    for (int e = 0; e < 64; ++e) {
        float s = conv_b[e];
#pragma unroll
        for (int j = 0; j < 4; ++j) {
            int tt = t - 3 + j;
            if (tt >= 0) s += conv_w[e * 4 + j] * pre[(size_t)(n - 3 + j) * 64 + e];
        }
        s = s / (1.f + __expf(-s));
        o[64 + e] = (_Float16)s;
    }
    const int offs[4] = {1, 2, 4, 8};
#pragma unroll
    for (int i = 0; i < 4; ++i) {
        int k = offs[i];
        o[128 + i] = (_Float16)((t >= k && ch == chars[n - k]) ? 1.f : 0.f);
    }
#pragma unroll
    for (int i = 132; i < 160; ++i) o[i] = (_Float16)0.f;
}

// ---------------------------------------------------------------------------
// TinyScan: per (b,s) channel, s_t = max(sigmoid(g),1e-6)*s_{t-1} + d_t.
// 256 independent sequences; writes f16 states ready for the Wo GEMM.
// ---------------------------------------------------------------------------
__global__ void scan_kernel(const float* __restrict__ rawGI,
                            const float* __restrict__ bg,
                            const float* __restrict__ bi,
                            _Float16* __restrict__ states, int T)
{
    int b = blockIdx.x, s = threadIdx.x;  // gridDim.x = B, blockDim.x = 32
    const float* base = rawGI + (size_t)b * T * 64;
    _Float16* sb = states + (size_t)b * T * 32;
    float bgv = bg[s], biv = bi[s], st = 0.f;
    for (int t = 0; t < T; ++t) {
        float g = 1.f / (1.f + __expf(-(base[(size_t)t * 64 + s] + bgv)));
        float d = (1.f - g) * (base[(size_t)t * 64 + 32 + s] + biv);
        st = fmaxf(g, 1e-6f) * st + d;
        sb[(size_t)t * 32 + s] = (_Float16)st;
    }
}

// ---------------------------------------------------------------------------
// Row LayerNorm over H=512 (in place), refresh f16 copy.
// ---------------------------------------------------------------------------
__global__ __launch_bounds__(256) void ln_kernel(float* __restrict__ h,
                                                 _Float16* __restrict__ hh,
                                                 const float* __restrict__ g,
                                                 const float* __restrict__ bta)
{
    __shared__ float s1[256];
    __shared__ float s2[256];
    int row = blockIdx.x, tid = threadIdx.x;
    float* x = h + (size_t)row * 512;
    float a = x[tid], b = x[tid + 256];
    s1[tid] = a + b;
    s2[tid] = a * a + b * b;
    __syncthreads();
    for (int st = 128; st > 0; st >>= 1) {
        if (tid < st) { s1[tid] += s1[tid + st]; s2[tid] += s2[tid + st]; }
        __syncthreads();
    }
    float mu  = s1[0] * (1.f / 512.f);
    float var = s2[0] * (1.f / 512.f) - mu * mu;
    float r   = rsqrtf(var + 1e-5f);
    float y0 = (a - mu) * r * g[tid] + bta[tid];
    float y1 = (b - mu) * r * g[tid + 256] + bta[tid + 256];
    x[tid] = y0; x[tid + 256] = y1;
    _Float16* xh = hh + (size_t)row * 512;
    xh[tid] = (_Float16)y0; xh[tid + 256] = (_Float16)y1;
}

// ---------------------------------------------------------------------------
// Helpers: f32 -> f16 conversion (with zero row padding) and Wg|Wi packing.
// ---------------------------------------------------------------------------
__global__ void cvt_pad_kernel(const float* __restrict__ src, _Float16* __restrict__ dst,
                               int srcRows, int cols, int dstRows)
{
    int i = blockIdx.x * 256 + threadIdx.x;
    int tot = dstRows * cols;
    if (i >= tot) return;
    int r = i / cols;
    dst[i] = (r < srcRows) ? (_Float16)src[i] : (_Float16)0.f;
}

__global__ void pack_gi_kernel(const float* __restrict__ Wg, const float* __restrict__ Wi,
                               _Float16* __restrict__ dst)
{
    int i = blockIdx.x * 256 + threadIdx.x;   // 512*64 total
    if (i >= 512 * 64) return;
    int k = i >> 6, s = i & 63;
    float v = (s < 32) ? Wg[k * 32 + s] : Wi[k * 32 + (s - 32)];
    dst[i] = (_Float16)v;
}

__global__ void cvt_f16_kernel(const float* __restrict__ src, _Float16* __restrict__ dst, int n)
{
    int i = blockIdx.x * 256 + threadIdx.x;
    if (i < n) dst[i] = (_Float16)src[i];
}

// ---------------------------------------------------------------------------
extern "C" void kernel_launch(void* const* d_in, const int* in_sizes, int n_in,
                              void* d_out, int out_size, void* d_ws, size_t ws_size,
                              hipStream_t stream)
{
    (void)in_sizes; (void)n_in; (void)out_size; (void)ws_size;
    const int B = 8, T = 8192, M = B * T;

    const int*   chars    = (const int*)  d_in[0];
    const float* emb_byte = (const float*)d_in[1];
    const float* hash_tb  = (const float*)d_in[2];
    const float* Wq       = (const float*)d_in[3];
    const float* bq       = (const float*)d_in[4];
    const float* conv_w   = (const float*)d_in[5];
    const float* conv_b   = (const float*)d_in[6];
    const float* Win      = (const float*)d_in[7];
    const float* b_in     = (const float*)d_in[8];
    const float* Wg       = (const float*)d_in[9];
    const float* bg       = (const float*)d_in[10];
    const float* Wi       = (const float*)d_in[11];
    const float* bi       = (const float*)d_in[12];
    const float* Wo       = (const float*)d_in[13];
    const float* bo       = (const float*)d_in[14];
    const float* ln_g     = (const float*)d_in[15];
    const float* ln_b     = (const float*)d_in[16];
    const float* mlp_w    = (const float*)d_in[17];
    const float* mlp_b    = (const float*)d_in[18];
    const float* Wout     = (const float*)d_in[19];
    const float* bout     = (const float*)d_in[20];

    // Workspace carve (256B-aligned chunks)
    char* ws = (char*)d_ws;
    size_t off = 0;
    auto carve = [&](size_t bytes) -> char* {
        char* p = ws + off;
        off = (off + bytes + 255) & ~(size_t)255;
        return p;
    };
    float*     pre    = (float*)    carve((size_t)M * 64 * 4);      // pre-conv hash feat
    _Float16*  feat   = (_Float16*) carve((size_t)M * 160 * 2);     // padded feature rows
    float*     hf     = (float*)    carve((size_t)M * 512 * 4);     // h (f32)
    _Float16*  hh     = (_Float16*) carve((size_t)M * 512 * 2);     // h (f16)
    float*     rawGI  = (float*)    carve((size_t)M * 64 * 4);      // [h@Wg | h@Wi]
    _Float16*  states = (_Float16*) carve((size_t)M * 32 * 2);      // scan states (f16)
    _Float16*  Win_h  = (_Float16*) carve((size_t)160 * 512 * 2);
    _Float16*  Wgi_h  = (_Float16*) carve((size_t)512 * 64 * 2);
    _Float16*  Wo_h   = (_Float16*) carve((size_t)32 * 512 * 2);
    _Float16*  mlp_h  = (_Float16*) carve((size_t)3 * 512 * 512 * 2);
    _Float16*  Wout_h = (_Float16*) carve((size_t)512 * 256 * 2);

    // --- feature pipeline ---
    featurize_kernel<<<M / 256, 256, 0, stream>>>(chars, emb_byte, hash_tb, Wq, bq, pre, T);
    build_feat_kernel<<<M / 256, 256, 0, stream>>>(chars, emb_byte, pre, conv_w, conv_b, feat, T);

    // --- weight conversion to f16 ---
    cvt_pad_kernel<<<(160 * 512 + 255) / 256, 256, 0, stream>>>(Win, Win_h, 132, 512, 160);
    pack_gi_kernel<<<(512 * 64 + 255) / 256, 256, 0, stream>>>(Wg, Wi, Wgi_h);
    cvt_pad_kernel<<<(32 * 512 + 255) / 256, 256, 0, stream>>>(Wo, Wo_h, 32, 512, 32);
    cvt_pad_kernel<<<(3 * 512 * 512 + 255) / 256, 256, 0, stream>>>(mlp_w, mlp_h, 3 * 512, 512, 3 * 512);
    cvt_pad_kernel<<<(512 * 256 + 255) / 256, 256, 0, stream>>>(Wout, Wout_h, 512, 256, 512);

    // --- h = silu(feat @ Win + b_in) : M x 160 x 512 ---
    gemm_f16_wmma<<<dim3(M / BM, 512 / BN), 256, 0, stream>>>(
        feat, 160, Win_h, b_in, nullptr, hf, hh, 512, 160, 1);

    // --- rawGI = h @ [Wg|Wi] : M x 512 x 64 (biases folded into scan) ---
    gemm_f16_wmma<<<dim3(M / BM, 64 / BN), 256, 0, stream>>>(
        hh, 512, Wgi_h, nullptr, nullptr, rawGI, nullptr, 64, 512, 0);

    // --- sequential gated scan (B x S = 256 channels) ---
    scan_kernel<<<B, 32, 0, stream>>>(rawGI, bg, bi, states, T);

    // --- h += states @ Wo + bo : M x 32 x 512, then LayerNorm ---
    gemm_f16_wmma<<<dim3(M / BM, 512 / BN), 256, 0, stream>>>(
        states, 32, Wo_h, bo, hf, hf, nullptr, 512, 32, 0);
    ln_kernel<<<M, 256, 0, stream>>>(hf, hh, ln_g, ln_b);

    // --- 3 residual MLP blocks: h += silu(h @ W + b) ---
    for (int l = 0; l < 3; ++l) {
        gemm_f16_wmma<<<dim3(M / BM, 512 / BN), 256, 0, stream>>>(
            hh, 512, mlp_h + (size_t)l * 512 * 512, mlp_b + (size_t)l * 512,
            hf, hf, nullptr, 512, 512, 1);
        cvt_f16_kernel<<<(M * 512 + 255) / 256, 256, 0, stream>>>(hf, hh, M * 512);
    }

    // --- logits = h @ Wout + bout : M x 512 x 256 -> d_out ---
    gemm_f16_wmma<<<dim3(M / BM, 256 / BN), 256, 0, stream>>>(
        hh, 512, Wout_h, bout, nullptr, (float*)d_out, nullptr, 256, 512, 0);
}
